// CausalSelfAttention_RoPE_25142738551001
// MI455X (gfx1250) — compile-verified, matrix operand order checked by reading, not run on hardware
//
#include <hip/hip_runtime.h>
#include <math.h>

typedef _Float16 h16;
typedef __attribute__((ext_vector_type(8)))  _Float16 v8h;
typedef __attribute__((ext_vector_type(16))) _Float16 v16h;
typedef __attribute__((ext_vector_type(8)))  float    v8f;

#define T_SEQ 4096
#define C_DIM 1024
#define H_NUM 16
#define HD    64

// ---------------------------------------------------------------------------
// Fragment helpers. CDNA5 16x16x32 f16 WMMA: per lane, A holds row (lane&15),
// with K chunks {half*8..half*8+7} and {16+half*8..}, half = lane>>4.
// B is symmetric per-lane over its column. Both are 16 halves = v16h.
// ---------------------------------------------------------------------------
__device__ __forceinline__ v16h frag_combine(v8h lo, v8h hi) {
  return __builtin_shufflevector(lo, hi, 0, 1, 2, 3, 4, 5, 6, 7,
                                          8, 9, 10, 11, 12, 13, 14, 15);
}

__device__ __forceinline__ v16h load_frag(const h16* p) {
  v8h lo = *(const v8h*)(p);
  v8h hi = *(const v8h*)(p + 16);
  return frag_combine(lo, hi);
}

__device__ __forceinline__ v8f wmma_f16(v16h a, v16h b, v8f c) {
  return __builtin_amdgcn_wmma_f32_16x16x32_f16(false, a, false, b,
                                                (short)0, c, false, false);
}

// ---------------------------------------------------------------------------
// f32 -> f16 convert
// ---------------------------------------------------------------------------
__global__ void f32_to_f16_kernel(const float* __restrict__ in,
                                  h16* __restrict__ out, int n) {
  int i = blockIdx.x * blockDim.x + threadIdx.x;
  if (i < n) out[i] = (h16)in[i];
}

// ---------------------------------------------------------------------------
// GEMM: C[M,N] = A[M,K] * W[N,K]^T, K == 1024 (contraction contiguous in both)
// Block: 256 thr = 8 waves; block tile 128(M) x 16(N); LDS-staged W tile.
// ---------------------------------------------------------------------------
template <bool STORE_HALF>
__global__ __launch_bounds__(256) void gemm_wmma_kernel(
    const h16* __restrict__ A, const h16* __restrict__ W,
    void* __restrict__ Cout, int M, int N, int K) {
  __shared__ h16 sB[16 * 1024];

  const int tid    = threadIdx.x;
  const int wave   = tid >> 5;
  const int lane   = tid & 31;
  const int rowm   = lane & 15;
  const int halfid = lane >> 4;
  const int n0     = blockIdx.y * 16;

  // Cooperative stage of 16 weight rows (16 x 1024 halves = 32 KB) into LDS.
  for (int i = tid; i < 16 * 128; i += 256) {
    int r = i >> 7;
    int c = (i & 127) * 8;
    *(v8h*)&sB[r * 1024 + c] = *(const v8h*)(W + (size_t)(n0 + r) * K + c);
  }
  __syncthreads();

  const int  mrow = blockIdx.x * 128 + wave * 16 + rowm;
  const h16* arow = A + (size_t)mrow * K;

  v8f acc = {};
  for (int k0 = 0; k0 < 1024; k0 += 32) {
    v16h a = load_frag(arow + k0 + halfid * 8);
    v16h b = load_frag(&sB[rowm * 1024 + k0 + halfid * 8]);
    acc = wmma_f16(a, b, acc);
  }

  // C layout: VGPR r, lanes 0-15 -> row r, lanes 16-31 -> row r+8; col = lane&15
  const int m0   = blockIdx.x * 128 + wave * 16 + 8 * halfid;
  const int ncol = n0 + rowm;
  if (STORE_HALF) {
    h16* Ch = (h16*)Cout;
#pragma unroll
    for (int r = 0; r < 8; ++r)
      Ch[(size_t)(m0 + r) * N + ncol] = (h16)acc[r];
  } else {
    float* Cf = (float*)Cout;
#pragma unroll
    for (int r = 0; r < 8; ++r)
      Cf[(size_t)(m0 + r) * N + ncol] = acc[r];
  }
}

// ---------------------------------------------------------------------------
// RoPE + arrange: qkv[T,3072] (f16) -> q[H][T][64], k[H][T][64] (rotated),
// v transposed to vT[H][64][T] so P*V B-fragments are contiguous loads.
// One thread per (t, h, p) with p = rotation-pair index (hd/2 = 32).
// ---------------------------------------------------------------------------
__global__ void rope_arrange_kernel(const h16* __restrict__ qkv,
                                    const int* __restrict__ pos,
                                    h16* __restrict__ qb,
                                    h16* __restrict__ kb,
                                    h16* __restrict__ vtb) {
  int i = blockIdx.x * blockDim.x + threadIdx.x;
  if (i >= T_SEQ * H_NUM * (HD / 2)) return;
  int p = i & 31;
  int h = (i >> 5) & 15;
  int t = i >> 9;

  float fpos = (float)pos[t];
  float invf = __powf(10000.0f, -(2.0f * (float)p) / 64.0f);
  float ang  = fpos * invf;
  float c    = __cosf(ang);
  float s    = __sinf(ang);

  const h16* row = qkv + (size_t)t * 3072;
  const int  d0  = h * 64 + 2 * p;

  size_t oidx = ((size_t)h * T_SEQ + t) * 64 + 2 * p;

  float qe = (float)row[d0];
  float qo = (float)row[d0 + 1];
  qb[oidx]     = (h16)(qe * c - qo * s);
  qb[oidx + 1] = (h16)(qe * s + qo * c);

  float ke = (float)row[1024 + d0];
  float ko = (float)row[1024 + d0 + 1];
  kb[oidx]     = (h16)(ke * c - ko * s);
  kb[oidx + 1] = (h16)(ke * s + ko * c);

  vtb[((size_t)h * 64 + 2 * p) * T_SEQ + t]     = row[2048 + d0];
  vtb[((size_t)h * 64 + 2 * p + 1) * T_SEQ + t] = row[2048 + d0 + 1];
}

// ---------------------------------------------------------------------------
// Flash attention (causal, online softmax). 256 thr = 8 waves per block;
// each wave owns 16 query rows of one head. Key blocks of 32.
//   S(16x32)  : 4 x wmma (2 K-steps over hd, 2 key half-tiles)
//   P->A frag : per-wave LDS bounce (C layout -> A layout), s_wait_dscnt
//   O += P*V  : 4 x wmma (4 d-chunks of 16), V read from vT[H][64][T]
// ---------------------------------------------------------------------------
__global__ __launch_bounds__(256) void flash_attn_kernel(
    const h16* __restrict__ qb, const h16* __restrict__ kb,
    const h16* __restrict__ vtb, h16* __restrict__ ob) {
  __shared__ h16 lds_p[8][16][32];  // per-wave P tile (16 rows x 32 keys)

  const int tid    = threadIdx.x;
  const int wave   = tid >> 5;
  const int lane   = tid & 31;
  const int rowm   = lane & 15;
  const int halfid = lane >> 4;
  const int row_hi = 8 * halfid;

  const int head = blockIdx.x >> 5;  // T/128 = 32 query blocks
  const int qblk = blockIdx.x & 31;
  const int t0   = qblk * 128 + wave * 16;

  const h16* qh = qb  + (size_t)head * T_SEQ * 64;
  const h16* kh = kb  + (size_t)head * T_SEQ * 64;
  const h16* vh = vtb + (size_t)head * 64 * T_SEQ;

  v16h qf0 = load_frag(qh + (size_t)(t0 + rowm) * 64 + 0  + halfid * 8);
  v16h qf1 = load_frag(qh + (size_t)(t0 + rowm) * 64 + 32 + halfid * 8);

  v8f o0 = {}, o1 = {}, o2 = {}, o3 = {};
  float mrun[8], lrun[8];
#pragma unroll
  for (int r = 0; r < 8; ++r) { mrun[r] = -1e30f; lrun[r] = 0.0f; }

  const float scale = 0.125f;  // 1/sqrt(64)

  for (int j0 = 0; j0 <= t0 + 15; j0 += 32) {
    asm volatile("" ::: "memory");  // keep LDS stores below prior P loads

    // ---- S = Q * K^T for 32 keys ----
    v8f s0 = {}, s1 = {};
    {
      v16h b00 = load_frag(kh + (size_t)(j0 + rowm) * 64 + 0  + halfid * 8);
      v16h b01 = load_frag(kh + (size_t)(j0 + rowm) * 64 + 32 + halfid * 8);
      s0 = wmma_f16(qf0, b00, s0);
      s0 = wmma_f16(qf1, b01, s0);
      v16h b10 = load_frag(kh + (size_t)(j0 + 16 + rowm) * 64 + 0  + halfid * 8);
      v16h b11 = load_frag(kh + (size_t)(j0 + 16 + rowm) * 64 + 32 + halfid * 8);
      s1 = wmma_f16(qf0, b10, s1);
      s1 = wmma_f16(qf1, b11, s1);
    }

    // ---- scale + causal mask + row max ----
    const int key0 = j0 + rowm;
    const int key1 = j0 + 16 + rowm;
    float mx[8];
#pragma unroll
    for (int r = 0; r < 8; ++r) {
      const int qrow = t0 + r + row_hi;
      float a = s0[r] * scale;
      float b = s1[r] * scale;
      if (key0 > qrow) a = -1e30f;
      if (key1 > qrow) b = -1e30f;
      s0[r] = a;
      s1[r] = b;
      mx[r] = fmaxf(a, b);
    }
#pragma unroll
    for (int r = 0; r < 8; ++r) {
      mx[r] = fmaxf(mx[r], __shfl_xor(mx[r], 1, 32));
      mx[r] = fmaxf(mx[r], __shfl_xor(mx[r], 2, 32));
      mx[r] = fmaxf(mx[r], __shfl_xor(mx[r], 4, 32));
      mx[r] = fmaxf(mx[r], __shfl_xor(mx[r], 8, 32));
    }

    // ---- online softmax update ----
    float rs[8];
#pragma unroll
    for (int r = 0; r < 8; ++r) {
      float mnew  = fmaxf(mrun[r], mx[r]);
      float p0    = __expf(s0[r] - mnew);
      float p1    = __expf(s1[r] - mnew);
      float alpha = __expf(mrun[r] - mnew);
      s0[r] = p0;
      s1[r] = p1;
      rs[r] = p0 + p1;
      mrun[r] = mnew;
      lrun[r] *= alpha;
      o0[r] *= alpha; o1[r] *= alpha; o2[r] *= alpha; o3[r] *= alpha;
    }
#pragma unroll
    for (int r = 0; r < 8; ++r) {
      rs[r] += __shfl_xor(rs[r], 1, 32);
      rs[r] += __shfl_xor(rs[r], 2, 32);
      rs[r] += __shfl_xor(rs[r], 4, 32);
      rs[r] += __shfl_xor(rs[r], 8, 32);
      lrun[r] += rs[r];
    }

    // ---- P: C layout -> A fragment via per-wave LDS ----
#pragma unroll
    for (int r = 0; r < 8; ++r) {
      lds_p[wave][r + row_hi][rowm]      = (h16)s0[r];
      lds_p[wave][r + row_hi][16 + rowm] = (h16)s1[r];
    }
    asm volatile("s_wait_dscnt 0" ::: "memory");
    v16h pf = load_frag(&lds_p[wave][rowm][halfid * 8]);

    // ---- O += P * V (V via vT: lane column = d, K = keys, contiguous) ----
    v16h bv0 = load_frag(vh + (size_t)(0  + rowm) * T_SEQ + j0 + halfid * 8);
    v16h bv1 = load_frag(vh + (size_t)(16 + rowm) * T_SEQ + j0 + halfid * 8);
    v16h bv2 = load_frag(vh + (size_t)(32 + rowm) * T_SEQ + j0 + halfid * 8);
    v16h bv3 = load_frag(vh + (size_t)(48 + rowm) * T_SEQ + j0 + halfid * 8);
    o0 = wmma_f16(pf, bv0, o0);
    o1 = wmma_f16(pf, bv1, o1);
    o2 = wmma_f16(pf, bv2, o2);
    o3 = wmma_f16(pf, bv3, o3);
  }

  // ---- epilogue: normalize and store to attn buffer [T][C] ----
#pragma unroll
  for (int r = 0; r < 8; ++r) {
    float  inv  = 1.0f / lrun[r];
    size_t base = (size_t)(t0 + r + row_hi) * C_DIM + head * 64;
    ob[base + 0  + rowm] = (h16)(o0[r] * inv);
    ob[base + 16 + rowm] = (h16)(o1[r] * inv);
    ob[base + 32 + rowm] = (h16)(o2[r] * inv);
    ob[base + 48 + rowm] = (h16)(o3[r] * inv);
  }
}

// ---------------------------------------------------------------------------
// Host-side orchestration (graph-capture safe: only kernel launches on stream)
// ---------------------------------------------------------------------------
extern "C" void kernel_launch(void* const* d_in, const int* in_sizes, int n_in,
                              void* d_out, int out_size, void* d_ws,
                              size_t ws_size, hipStream_t stream) {
  const float* x     = (const float*)d_in[0];
  const float* w_qkv = (const float*)d_in[1];
  const float* w_out = (const float*)d_in[2];
  const int*   tok   = (const int*)d_in[3];

  char* ws = (char*)d_ws;
  h16* x16    = (h16*)(ws + 0);               // 4096*1024  (8 MiB)
  h16* wqkv16 = (h16*)(ws + 8388608);         // 3072*1024  (6 MiB)
  h16* wout16 = (h16*)(ws + 14680064);        // 1024*1024  (2 MiB)
  h16* qkv16  = (h16*)(ws + 16777216);        // 4096*3072  (24 MiB)
  h16* qb     = (h16*)(ws + 41943040);        // 16*4096*64 (8 MiB)
  h16* kb     = (h16*)(ws + 50331648);        // 16*4096*64 (8 MiB)
  h16* vtb    = (h16*)(ws + 58720256);        // 16*64*4096 (8 MiB)
  h16* attn16 = (h16*)(ws + 67108864);        // 4096*1024  (8 MiB)

  int n;
  n = T_SEQ * C_DIM;
  f32_to_f16_kernel<<<(n + 255) / 256, 256, 0, stream>>>(x, x16, n);
  n = 3 * C_DIM * C_DIM;
  f32_to_f16_kernel<<<(n + 255) / 256, 256, 0, stream>>>(w_qkv, wqkv16, n);
  n = C_DIM * C_DIM;
  f32_to_f16_kernel<<<(n + 255) / 256, 256, 0, stream>>>(w_out, wout16, n);

  // qkv = x * w_qkv^T : (4096 x 1024) x (1024 x 3072)
  gemm_wmma_kernel<true><<<dim3(T_SEQ / 128, 3 * C_DIM / 16), 256, 0, stream>>>(
      x16, wqkv16, qkv16, T_SEQ, 3 * C_DIM, C_DIM);

  // RoPE + head-major arrange (+ V transpose)
  n = T_SEQ * H_NUM * (HD / 2);
  rope_arrange_kernel<<<(n + 255) / 256, 256, 0, stream>>>(qkv16, tok, qb, kb,
                                                           vtb);

  // Flash attention: H * (T/128) blocks
  flash_attn_kernel<<<H_NUM * (T_SEQ / 128), 256, 0, stream>>>(qb, kb, vtb,
                                                               attn16);

  // out = attn * w_out^T : (4096 x 1024) x (1024 x 1024) -> f32
  gemm_wmma_kernel<false><<<dim3(T_SEQ / 128, C_DIM / 16), 256, 0, stream>>>(
      attn16, wout16, d_out, T_SEQ, C_DIM, C_DIM);
}
